// DenseFlashAttention_82617990905949
// MI455X (gfx1250) — compile-verified, matrix operand order checked by reading, not run on hardware
//
#include <hip/hip_runtime.h>
#include <hip/hip_bf16.h>

// GraphAttention for MI455X (gfx1250): bf16x3 error-compensated WMMA GEMMs +
// L2-resident wave-per-edge gather/scatter softmax.

typedef __attribute__((ext_vector_type(16))) __bf16 v16bf;
typedef __attribute__((ext_vector_type(8)))  float  v8f;

union Frag16 { v16bf v; uint4 u[2]; };

static __device__ __forceinline__ v8f wmma_bf16(v16bf a, v16bf b, v8f c) {
  // (neg_a, A, neg_b, B, c_mod, C, reuse_a, reuse_b)
  return __builtin_amdgcn_wmma_f32_16x16x32_bf16(false, a, false, b, (short)0, c, false, false);
}

static __device__ __forceinline__ void atomicMaxF(float* addr, float val) {
  if (val >= 0.0f) atomicMax((int*)addr, __float_as_int(val));
  else             atomicMin((unsigned int*)addr, __float_as_uint(val));
}

__global__ void ga_init_kernel(float* __restrict__ numer, float* __restrict__ denom,
                               float* __restrict__ m, int n, int nd) {
  int i = blockIdx.x * blockDim.x + threadIdx.x;
  if (i < nd) numer[i] = 0.0f;
  if (i < n) { denom[i] = 0.0f; m[i] = -__builtin_inff(); }
}

__global__ void ga_split_kernel(const float* __restrict__ src,
                                __bf16* __restrict__ hi, __bf16* __restrict__ lo, int n) {
  int i = blockIdx.x * blockDim.x + threadIdx.x;
  if (i >= n) return;
  float v = src[i];
  __bf16 h = (__bf16)v;
  hi[i] = h;
  lo[i] = (__bf16)(v - (float)h);
}

__global__ void ga_wsplit_t_kernel(const float* __restrict__ W,
                                   __bf16* __restrict__ hi, __bf16* __restrict__ lo) {
  int i = blockIdx.x * blockDim.x + threadIdx.x;   // 0..16383
  int k = i >> 7, n = i & 127;
  float v = W[i];
  __bf16 h = (__bf16)v;
  hi[n * 128 + k] = h;
  lo[n * 128 + k] = (__bf16)(v - (float)h);
}

// grid = (nrows/16, 1, 3), block = 256 (8 waves). Wave w -> column tile w.
__global__ __launch_bounds__(256) void ga_gemm_qkv_kernel(
    const __bf16* __restrict__ xhi, const __bf16* __restrict__ xlo,
    const __bf16* __restrict__ whi, const __bf16* __restrict__ wlo,
    float* __restrict__ qkv, int nrows) {
  const int lane = threadIdx.x & 31;
  const int wave = threadIdx.x >> 5;
  const int half = lane >> 4;
  const int l16  = lane & 15;
  const int rowBase = blockIdx.x * 16;
  const int colBase = wave * 16;
  int arow = rowBase + l16; if (arow >= nrows) arow = nrows - 1;

  const uint4* Ah = (const uint4*)(xhi + (size_t)arow * 128);
  const uint4* Al = (const uint4*)(xlo + (size_t)arow * 128);
  const size_t wbase = (size_t)blockIdx.z * 16384 + (size_t)(colBase + l16) * 128;
  const uint4* Bh = (const uint4*)(whi + wbase);
  const uint4* Bl = (const uint4*)(wlo + wbase);

  v8f c = {};
#pragma unroll
  for (int k0 = 0; k0 < 128; k0 += 32) {
    const int cu = (k0 >> 3) + half;
    Frag16 ah, al, bh, bl;
    ah.u[0] = Ah[cu]; ah.u[1] = Ah[cu + 2];
    al.u[0] = Al[cu]; al.u[1] = Al[cu + 2];
    bh.u[0] = Bh[cu]; bh.u[1] = Bh[cu + 2];
    bl.u[0] = Bl[cu]; bl.u[1] = Bl[cu + 2];
    c = wmma_bf16(ah.v, bh.v, c);
    c = wmma_bf16(ah.v, bl.v, c);
    c = wmma_bf16(al.v, bh.v, c);
  }
  float* outm = qkv + (size_t)blockIdx.z * (size_t)nrows * 128;
#pragma unroll
  for (int r = 0; r < 8; ++r) {
    int mrow = rowBase + r + 8 * half;
    if (mrow < nrows) outm[(size_t)mrow * 128 + colBase + l16] = c[r];
  }
}

__global__ void ga_edge_score_kernel(const int* __restrict__ ei,
                                     const float* __restrict__ Q,
                                     const float* __restrict__ K,
                                     float* __restrict__ esc, float* __restrict__ m,
                                     int nedges, float scale) {
  int edge = (int)((blockIdx.x * (unsigned)blockDim.x + threadIdx.x) >> 5);
  int lane = threadIdx.x & 31;
  if (edge >= nedges) return;
  int s = ei[edge];
  int r = ei[nedges + edge];
  float4 a = ((const float4*)(Q + (size_t)r * 128))[lane];
  float4 b = ((const float4*)(K + (size_t)s * 128))[lane];
  float d = a.x * b.x + a.y * b.y + a.z * b.z + a.w * b.w;
#pragma unroll
  for (int off = 16; off > 0; off >>= 1) d += __shfl_xor(d, off, 32);
  if (lane == 0) {
    float sc = d * scale;
    esc[edge] = sc;
    atomicMaxF(&m[r], sc);
  }
}

__global__ void ga_fix_m_kernel(float* __restrict__ m, int n) {
  int i = blockIdx.x * blockDim.x + threadIdx.x;
  if (i < n && !isfinite(m[i])) m[i] = 0.0f;
}

__global__ void ga_edge_accum_kernel(const int* __restrict__ ei,
                                     const float* __restrict__ esc,
                                     const float* __restrict__ m,
                                     const float* __restrict__ V,
                                     float* __restrict__ denom,
                                     float* __restrict__ numer, int nedges) {
  int edge = (int)((blockIdx.x * (unsigned)blockDim.x + threadIdx.x) >> 5);
  int lane = threadIdx.x & 31;
  if (edge >= nedges) return;
  int s = ei[edge];
  int r = ei[nedges + edge];
  float w = expf(esc[edge] - m[r]);
  if (lane == 0) atomicAdd(&denom[r], w);
  float4 v = ((const float4*)(V + (size_t)s * 128))[lane];
  float* nb = numer + (size_t)r * 128 + lane * 4;
  atomicAdd(nb + 0, w * v.x);
  atomicAdd(nb + 1, w * v.y);
  atomicAdd(nb + 2, w * v.z);
  atomicAdd(nb + 3, w * v.w);
}

__global__ __launch_bounds__(256) void ga_gemm_out_kernel(
    const float* __restrict__ numer, const float* __restrict__ denom,
    const float* __restrict__ x,
    const __bf16* __restrict__ bhi, const __bf16* __restrict__ blo,
    float* __restrict__ out, int nrows) {
  const int lane = threadIdx.x & 31;
  const int wave = threadIdx.x >> 5;
  const int half = lane >> 4;
  const int l16  = lane & 15;
  const int rowBase = blockIdx.x * 16;
  const int colBase = wave * 16;
  int arow = rowBase + l16; if (arow >= nrows) arow = nrows - 1;

  const float dn  = denom[arow];
  const float inv = dn > 0.0f ? 1.0f / dn : 0.0f;
  const float4* A4 = (const float4*)(numer + (size_t)arow * 128);
  const uint4*  Bh = (const uint4*)(bhi + (size_t)(colBase + l16) * 128);
  const uint4*  Bl = (const uint4*)(blo + (size_t)(colBase + l16) * 128);

  v8f c = {};
#pragma unroll
  for (int k0 = 0; k0 < 128; k0 += 32) {
    const int kA = (k0 + 8 * half) >> 2;
    const int kB = (k0 + 16 + 8 * half) >> 2;
    float4 f0 = A4[kA], f1 = A4[kA + 1], f2 = A4[kB], f3 = A4[kB + 1];
    float vals[16] = { f0.x, f0.y, f0.z, f0.w, f1.x, f1.y, f1.z, f1.w,
                       f2.x, f2.y, f2.z, f2.w, f3.x, f3.y, f3.z, f3.w };
    Frag16 ah, al, bh, bl;
#pragma unroll
    for (int j = 0; j < 16; ++j) {
      float f = vals[j] * inv;
      __bf16 h = (__bf16)f;
      ah.v[j] = h;
      al.v[j] = (__bf16)(f - (float)h);
    }
    const int cu = (k0 >> 3) + half;
    bh.u[0] = Bh[cu]; bh.u[1] = Bh[cu + 2];
    bl.u[0] = Bl[cu]; bl.u[1] = Bl[cu + 2];
    c = wmma_bf16(ah.v, bh.v, c);
    c = wmma_bf16(ah.v, bl.v, c);
    c = wmma_bf16(al.v, bh.v, c);
  }
#pragma unroll
  for (int r = 0; r < 8; ++r) {
    int mrow = rowBase + r + 8 * half;
    if (mrow < nrows) {
      size_t o = (size_t)mrow * 128 + colBase + l16;
      out[o] = x[o] + c[r];
    }
  }
}

extern "C" void kernel_launch(void* const* d_in, const int* in_sizes, int n_in,
                              void* d_out, int out_size, void* d_ws, size_t ws_size,
                              hipStream_t stream) {
  const float* x  = (const float*)d_in[0];
  const float* Wq = (const float*)d_in[1];
  const float* Wk = (const float*)d_in[2];
  const float* Wv = (const float*)d_in[3];
  const float* Wo = (const float*)d_in[4];
  const int*   ei = (const int*)d_in[5];      // [2][E]: row0 sender, row1 receiver
  float* out = (float*)d_out;

  const int D  = 128;
  const int N  = in_sizes[0] / D;             // 40000
  const int E  = in_sizes[5] / 2;             // 640000
  const size_t ND = (size_t)N * D;

  char* p = (char*)d_ws;
  float*  qkv   = (float*)p;  p += 3 * ND * 4;
  float*  numer = (float*)p;  p += ND * 4;
  float*  esc   = (float*)p;  p += (size_t)E * 4;
  float*  m     = (float*)p;  p += (size_t)N * 4;
  float*  denom = (float*)p;  p += (size_t)N * 4;
  __bf16* xhi   = (__bf16*)p; p += ND * 2;
  __bf16* xlo   = (__bf16*)p; p += ND * 2;
  __bf16* whi   = (__bf16*)p; p += (size_t)4 * 16384 * 2;
  __bf16* wlo   = (__bf16*)p;

  float* Q  = qkv;
  float* Kp = qkv + ND;
  float* V  = qkv + 2 * ND;

  const float scale = 0.08838834764831845f;   // 1/sqrt(128)
  const int tb = 256;
  const int gND = (int)((ND + tb - 1) / tb);

  ga_init_kernel<<<gND, tb, 0, stream>>>(numer, denom, m, N, (int)ND);
  ga_split_kernel<<<gND, tb, 0, stream>>>(x, xhi, xlo, (int)ND);
  ga_wsplit_t_kernel<<<64, tb, 0, stream>>>(Wq, whi + 0 * 16384, wlo + 0 * 16384);
  ga_wsplit_t_kernel<<<64, tb, 0, stream>>>(Wk, whi + 1 * 16384, wlo + 1 * 16384);
  ga_wsplit_t_kernel<<<64, tb, 0, stream>>>(Wv, whi + 2 * 16384, wlo + 2 * 16384);
  ga_wsplit_t_kernel<<<64, tb, 0, stream>>>(Wo, whi + 3 * 16384, wlo + 3 * 16384);

  ga_gemm_qkv_kernel<<<dim3((N + 15) / 16, 1, 3), tb, 0, stream>>>(xhi, xlo, whi, wlo, qkv, N);

  ga_edge_score_kernel<<<(E + 7) / 8, tb, 0, stream>>>(ei, Q, Kp, esc, m, E, scale);
  ga_fix_m_kernel<<<(N + tb - 1) / tb, tb, 0, stream>>>(m, N);
  ga_edge_accum_kernel<<<(E + 7) / 8, tb, 0, stream>>>(ei, esc, m, V, denom, numer, E);

  ga_gemm_out_kernel<<<(N + 15) / 16, tb, 0, stream>>>(numer, denom, x,
                                                       whi + 3 * 16384, wlo + 3 * 16384,
                                                       out, N);
}
